// GAT_23055384445458
// MI455X (gfx1250) — matrix operand order repaired and checked by reading, not compiled, not run on hardware
//
#include <hip/hip_runtime.h>
#include <hip/hip_bf16.h>
#include <math.h>

#define NN      2048
#define NWORD   64          // NN/32 mask words per row
#define NSTEPS  17          // prune loop runs i=0..16 (max_gat_iter=16)
#define T1R     0           // round(0.03*16)
#define T2R     1           // round(0.08*16)

typedef __attribute__((ext_vector_type(16))) _Float16 v16h;
typedef __attribute__((ext_vector_type(8)))  float    v8f;

__device__ __forceinline__ float elu1(float x)  { return x > 0.f ? x : expm1f(x); }
__device__ __forceinline__ float lrelu(float x) { return x > 0.f ? x : 0.2f * x; }

// ---------------------------------------------------------------- adjacency bits
__global__ void k_build_bits(const float* __restrict__ adj, unsigned* __restrict__ bits) {
    int t = blockIdx.x * blockDim.x + threadIdx.x;
    if (t >= NN * NWORD) return;
    int r = t >> 6, w = t & 63;
    const float* row = adj + (size_t)r * NN + w * 32;
    unsigned m = 0;
#pragma unroll
    for (int k = 0; k < 32; ++k)
        if (row[k] > 0.f) m |= (1u << k);
    bits[t] = m;
}

__global__ void k_degree(const unsigned* __restrict__ bits, int* __restrict__ deg) {
    int r = blockIdx.x * blockDim.x + threadIdx.x;
    if (r >= NN) return;
    int d = 0;
    for (int w = 0; w < NWORD; ++w) d += __popc(bits[(size_t)r * NWORD + w]);
    deg[r] = d;
}

__global__ void k_fill_ones(unsigned* __restrict__ w) {
    int t = threadIdx.x;
    if (t < NWORD) w[t] = 0xffffffffu;
}

// ---------------------------------------------------------------- prune schedule
// fp32 absorption: after any node dies, every alive degree == k + ndead*NEG and the
// small k is lost (ulp(9e15_f32) >> 2048), so min_d==max_d -> pop exactly one node
// (the tail of the INITIAL stable-descending sort) on every iteration i>=1.
__global__ void k_prune(const int* __restrict__ deg, int* __restrict__ ns,
                        int* __restrict__ alive, unsigned* __restrict__ alivew) {
    if (threadIdx.x != 0 || blockIdx.x != 0) return;
    int maxd = 0;
    for (int i = 0; i < NN; ++i) { alive[i] = 1; if (deg[i] > maxd) maxd = deg[i]; }
    // stable counting sort, descending degree, ascending index within ties
    int cnt = 0;
    for (int d = maxd; d >= 0; --d)
        for (int i = 0; i < NN; ++i)
            if (deg[i] == d) ns[cnt++] = i;
    int n = NN;
    // iteration 0: true integer degrees
    int min_d = deg[ns[n - 1]];
    int max_d = deg[ns[0]];
    alive[ns[--n]] = 0;
    if (n > 0 && min_d != max_d)
        while (n > 0 && deg[ns[n - 1]] == min_d) alive[ns[--n]] = 0;
    for (int w = 0; w < NWORD; ++w) {
        unsigned mw = 0;
        for (int b = 0; b < 32; ++b) if (alive[w * 32 + b]) mw |= (1u << b);
        alivew[w] = mw;
    }
    // iterations 1..16: uniform contaminated degrees -> one pop each
    for (int it = 1; it < NSTEPS; ++it) {
        if (n > 1) alive[ns[--n]] = 0;
        for (int w = 0; w < NWORD; ++w) {
            unsigned mw = 0;
            for (int b = 0; b < 32; ++b) if (alive[w * 32 + b]) mw |= (1u << b);
            alivew[(size_t)it * NWORD + w] = mw;
        }
    }
}

// ---------------------------------------------------------------- GAT phase 1: Wh, src, dst
// Stores Wh transposed as f16 [H][16][NN] (cols >= FH zero-padded) for the WMMA B operand.
template <int FIN, int FH, int H>
__global__ void k_prep(const float* __restrict__ x, const float* __restrict__ W,
                       const float* __restrict__ a, _Float16* __restrict__ whT,
                       float* __restrict__ src, float* __restrict__ dst) {
    int t = blockIdx.x * blockDim.x + threadIdx.x;
    if (t >= H * NN) return;
    int h = t / NN, i = t % NN;
    const float* xr = x + (size_t)i * FIN;
    const float* Wh = W + (size_t)h * FIN * FH;
    float acc[FH];
#pragma unroll
    for (int c = 0; c < FH; ++c) acc[c] = 0.f;
    for (int f = 0; f < FIN; ++f) {
        float xv = xr[f];
#pragma unroll
        for (int c = 0; c < FH; ++c) acc[c] = fmaf(xv, Wh[f * FH + c], acc[c]);
    }
    const float* ah = a + (size_t)h * 2 * FH;
    float s = 0.f, d = 0.f;
#pragma unroll
    for (int c = 0; c < FH; ++c) { s = fmaf(acc[c], ah[c], s); d = fmaf(acc[c], ah[FH + c], d); }
    src[(size_t)h * NN + i] = s;
    dst[(size_t)h * NN + i] = d;
#pragma unroll
    for (int c = 0; c < 16; ++c)
        whT[((size_t)h * 16 + c) * NN + i] = (c < FH) ? (_Float16)acc[c] : (_Float16)0.f;
}

// ---------------------------------------------------------------- row max (no exp; lrelu monotone)
__global__ void k_rowmax(const unsigned* __restrict__ adjb, const unsigned* __restrict__ alw,
                         const float* __restrict__ src, const float* __restrict__ dst,
                         float* __restrict__ mrow, int* __restrict__ emp, int H) {
    int wid  = (blockIdx.x * blockDim.x + threadIdx.x) >> 5;
    int lane = threadIdx.x & 31;
    if (wid >= H * NN) return;
    int h = wid / NN, i = wid % NN;
    unsigned rowMask = ((alw[i >> 5] >> (i & 31)) & 1) ? 0xffffffffu : 0u;
    float md = -3.4e38f;
    int found = 0;
    const float* dh = dst + (size_t)h * NN;
    for (int w = 0; w < NWORD; ++w) {
        unsigned mw = adjb[(size_t)i * NWORD + w] & alw[w] & rowMask;
        if ((mw >> lane) & 1) { md = fmaxf(md, dh[w * 32 + lane]); found = 1; }
    }
#pragma unroll
    for (int off = 16; off >= 1; off >>= 1) {
        md = fmaxf(md, __shfl_xor(md, off, 32));
        found |= __shfl_xor(found, off, 32);
    }
    if (lane == 0) {
        mrow[(size_t)h * NN + i] = found ? lrelu(src[(size_t)h * NN + i] + md) : 0.f;
        emp[(size_t)h * NN + i]  = found ? 0 : 1;
    }
}

// ---------------------------------------------------------------- fused attention + aggregate (WMMA)
// One wave per (head, 16-row tile). Per 32-column chunk: build P tile in the f16 A
// layout (lane = row M, K split by lane-half), B = two 8-elem runs of whT in A's K order, one
// v_wmma_f32_16x16x32_f16. Softmax denominator accumulated scalar-side per lane.
__global__ void k_attn(const unsigned* __restrict__ adjb, const unsigned* __restrict__ alw,
                       const float* __restrict__ src, const float* __restrict__ dst,
                       const float* __restrict__ mrow, const int* __restrict__ emp,
                       const _Float16* __restrict__ whT, float* __restrict__ raw, int H) {
    int wid  = (blockIdx.x * blockDim.x + threadIdx.x) >> 5;
    int lane = threadIdx.x & 31;
    if (wid >= H * (NN / 16)) return;
    int h = wid / (NN / 16), tile = wid % (NN / 16);
    int half = lane >> 4, r = lane & 15;
    int i = tile * 16 + r;

    float src_r = src[(size_t)h * NN + i];
    float m_r   = mrow[(size_t)h * NN + i];
    int   empr  = emp[(size_t)h * NN + i];
    unsigned rowMask = ((alw[i >> 5] >> (i & 31)) & 1) ? 0xffffffffu : 0u;
    const unsigned* adjrow = adjb + (size_t)i * NWORD;

    const float*    dh   = dst + (size_t)h * NN;
    const _Float16* bptr = whT + ((size_t)h * 16 + (lane & 15)) * NN;   // B column = lane%16

    v8f C = {0.f, 0.f, 0.f, 0.f, 0.f, 0.f, 0.f, 0.f};
    float psum = 0.f;

    for (int j0 = 0; j0 < NN; j0 += 32) {
        unsigned mw = adjrow[j0 >> 5] & alw[j0 >> 5] & rowMask;   // unconditional, branch-free
        int b1 = j0 + 8 * half;        // A elems 0..7  -> K = e + 8*half
        int b2 = j0 + 16 + 8 * half;   // A elems 8..15 -> K = (e-8) + 16 + 8*half
        float4 da = *(const float4*)(dh + b1);
        float4 db = *(const float4*)(dh + b1 + 4);
        float4 dc = *(const float4*)(dh + b2);
        float4 dd = *(const float4*)(dh + b2 + 4);
        float dv0[8] = {da.x, da.y, da.z, da.w, db.x, db.y, db.z, db.w};
        float dv1[8] = {dc.x, dc.y, dc.z, dc.w, dd.x, dd.y, dd.z, dd.w};
        v16h A;
#pragma unroll
        for (int e = 0; e < 8; ++e) {
            int k1 = 8 * half + e;
            float t1 = lrelu(src_r + dv0[e]);
            float p1 = ((mw >> k1) & 1) ? __expf(t1 - m_r) : 0.f;
            p1 = empr ? 1.f : p1;
            psum += p1;
            A[e] = (_Float16)p1;
            int k2 = 16 + 8 * half + e;
            float t2 = lrelu(src_r + dv1[e]);
            float p2 = ((mw >> k2) & 1) ? __expf(t2 - m_r) : 0.f;
            p2 = empr ? 1.f : p2;
            psum += p2;
            A[8 + e] = (_Float16)p2;
        }
        v16h B;   // K = 8*half + e (e<8), 16 + 8*half + (e-8) (e>=8)
#pragma unroll
        for (int e = 0; e < 8; ++e) { B[e] = bptr[b1 + e]; B[8 + e] = bptr[b2 + e]; }
        C = __builtin_amdgcn_wmma_f32_16x16x32_f16(false, A, false, B, (short)0, C, false, false);
    }

    psum += __shfl_xor(psum, 16, 32);   // lane L now holds rowsum for row L%16
    float inv_r = 1.0f / psum;          // ONE divide per row, broadcast the reciprocal
#pragma unroll
    for (int v = 0; v < 8; ++v) {
        int M = v + 8 * half;           // C[v]: row M, col lane%16
        float invM = __shfl(inv_r, M, 32);
        raw[(((size_t)h * NN) + tile * 16 + M) * 16 + (lane & 15)] = C[v] * invM;
    }
}

// ---------------------------------------------------------------- head combine + ELU
__global__ void k_combine_concat(const float* __restrict__ raw, float* __restrict__ y, int H, int FH) {
    int t = blockIdx.x * blockDim.x + threadIdx.x;
    if (t >= NN * H * FH) return;
    int i = t / (H * FH), k = t % (H * FH);
    int h = k / FH, c = k % FH;
    y[(size_t)i * (H * FH) + k] = elu1(raw[((size_t)h * NN + i) * 16 + c]);
}

__global__ void k_combine_mean2(const float* __restrict__ raw, float* __restrict__ y) {
    int t = blockIdx.x * blockDim.x + threadIdx.x;
    if (t >= NN * 16) return;
    int i = t >> 4, c = t & 15;
    float v = 0.5f * (raw[(size_t)i * 16 + c] + raw[((size_t)NN + i) * 16 + c]);
    y[t] = elu1(v);
}

// ---------------------------------------------------------------- masked mean + FC chain
__global__ void k_final(const float* __restrict__ x, const unsigned* __restrict__ alw,
                        const float* __restrict__ fc0w, const float* __restrict__ fc0b,
                        const float* __restrict__ fc1w, const float* __restrict__ fc1b,
                        const float* __restrict__ fc2w, const float* __restrict__ fc2b,
                        float* __restrict__ out) {
    __shared__ float ssum[256];
    __shared__ int   scnt[16];
    __shared__ float mean[16], h0[16], h1[8];
    int t = threadIdx.x;
    int col = t & 15, grp = t >> 4;
    float s = 0.f; int c = 0;
    for (int i = grp; i < NN; i += 16) {
        if ((alw[i >> 5] >> (i & 31)) & 1) { s += x[(size_t)i * 16 + col]; ++c; }
    }
    ssum[t] = s;
    if (col == 0) scnt[grp] = c;
    __syncthreads();
    if (t < 16) {
        float tot = 0.f; int cnt = 0;
        for (int g = 0; g < 16; ++g) { tot += ssum[g * 16 + t]; cnt += scnt[g]; }
        mean[t] = tot / (float)cnt;
    }
    __syncthreads();
    if (t < 16) {
        float acc = fc0b[t];
        for (int k = 0; k < 16; ++k) acc = fmaf(mean[k], fc0w[t * 16 + k], acc);
        h0[t] = elu1(acc);
    }
    __syncthreads();
    if (t < 8) {
        float acc = fc1b[t];
        for (int k = 0; k < 16; ++k) acc = fmaf(h0[k], fc1w[t * 16 + k], acc);
        h1[t] = elu1(acc);
    }
    __syncthreads();
    if (t == 0) {
        float acc = fc2b[0];
        for (int k = 0; k < 8; ++k) acc = fmaf(h1[k], fc2w[k], acc);
        out[0] = elu1(acc);
    }
}

// ---------------------------------------------------------------- host driver
extern "C" void kernel_launch(void* const* d_in, const int* in_sizes, int n_in,
                              void* d_out, int out_size, void* d_ws, size_t ws_size,
                              hipStream_t stream) {
    (void)in_sizes; (void)n_in; (void)out_size; (void)ws_size;
    const float* feat = (const float*)d_in[0];
    const float* adj  = (const float*)d_in[1];
    const float* W1 = (const float*)d_in[3];  const float* a1 = (const float*)d_in[4];
    const float* W2 = (const float*)d_in[5];  const float* a2 = (const float*)d_in[6];
    const float* W3 = (const float*)d_in[7];  const float* a3 = (const float*)d_in[8];
    const float* W4 = (const float*)d_in[9];  const float* a4 = (const float*)d_in[10];
    const float* fc0w = (const float*)d_in[11]; const float* fc0b = (const float*)d_in[12];
    const float* fc1w = (const float*)d_in[13]; const float* fc1b = (const float*)d_in[14];
    const float* fc2w = (const float*)d_in[15]; const float* fc2b = (const float*)d_in[16];
    float* out = (float*)d_out;

    char* wsb = (char*)d_ws;
    size_t off = 0;
    auto alloc = [&](size_t b) -> char* {
        char* p = wsb + off;
        off = (off + b + 255) & ~(size_t)255;
        return p;
    };
    unsigned* adjb     = (unsigned*)alloc((size_t)NN * NWORD * 4);
    int*      deg      = (int*)alloc((size_t)NN * 4);
    int*      ns       = (int*)alloc((size_t)NN * 4);
    int*      aliveArr = (int*)alloc((size_t)NN * 4);
    unsigned* alivew   = (unsigned*)alloc((size_t)NSTEPS * NWORD * 4);
    unsigned* onesw    = (unsigned*)alloc((size_t)NWORD * 4);
    _Float16* whT      = (_Float16*)alloc((size_t)4 * 16 * NN * 2);
    float*    srcb     = (float*)alloc((size_t)4 * NN * 4);
    float*    dstb     = (float*)alloc((size_t)4 * NN * 4);
    float*    mrow     = (float*)alloc((size_t)4 * NN * 4);
    int*      emp      = (int*)alloc((size_t)4 * NN * 4);
    float*    raw      = (float*)alloc((size_t)4 * NN * 16 * 4);
    float*    xa       = (float*)alloc((size_t)NN * 16 * 4);
    float*    xb       = (float*)alloc((size_t)NN * 16 * 4);

    k_build_bits<<<(NN * NWORD + 255) / 256, 256, 0, stream>>>(adj, adjb);
    k_degree<<<(NN + 255) / 256, 256, 0, stream>>>(adjb, deg);
    k_fill_ones<<<1, 64, 0, stream>>>(onesw);
    k_prune<<<1, 32, 0, stream>>>(deg, ns, aliveArr, alivew);

    auto run_layer = [&](const float* xin, const float* W, const float* a,
                         int H, int FH, int FIN, const unsigned* alw,
                         bool concat, float* yout) {
        int threads = H * NN;
        if (FIN == 64)
            k_prep<64, 4, 4><<<(threads + 255) / 256, 256, 0, stream>>>(xin, W, a, whT, srcb, dstb);
        else if (FH == 8)
            k_prep<16, 8, 2><<<(threads + 255) / 256, 256, 0, stream>>>(xin, W, a, whT, srcb, dstb);
        else
            k_prep<16, 16, 2><<<(threads + 255) / 256, 256, 0, stream>>>(xin, W, a, whT, srcb, dstb);
        k_rowmax<<<H * NN / 8, 256, 0, stream>>>(adjb, alw, srcb, dstb, mrow, emp, H);
        k_attn<<<H * 16, 256, 0, stream>>>(adjb, alw, srcb, dstb, mrow, emp, whT, raw, H);
        if (concat)
            k_combine_concat<<<(NN * H * FH + 255) / 256, 256, 0, stream>>>(raw, yout, H, FH);
        else
            k_combine_mean2<<<(NN * 16 + 255) / 256, 256, 0, stream>>>(raw, yout);
    };

    float* xcur = xa;
    float* xnxt = xb;
    run_layer(feat, W1, a1, 4, 4, 64, onesw, true, xcur);
    for (int it = 0; it < NSTEPS; ++it) {
        const unsigned* alw = alivew + (size_t)it * NWORD;
        if (it <= T1R) {
            run_layer(xcur, W2, a2, 2, 8, 16, alw, true, xnxt);
            float* t = xcur; xcur = xnxt; xnxt = t;
        }
        if (it <= T2R) {
            run_layer(xcur, W3, a3, 2, 8, 16, alw, true, xnxt);
            float* t = xcur; xcur = xnxt; xnxt = t;
        }
        run_layer(xcur, W4, a4, 2, 16, 16, alw, false, xnxt);
        float* t = xcur; xcur = xnxt; xnxt = t;
    }
    k_final<<<1, 256, 0, stream>>>(xcur, alivew + (size_t)(NSTEPS - 1) * NWORD,
                                   fc0w, fc0b, fc1w, fc1b, fc2w, fc2b, out);
}
